// FP8_Linear_81741817577923
// MI455X (gfx1250) — matrix operand order repaired and checked by reading, not compiled, not run on hardware
//
#include <hip/hip_runtime.h>
#include <hip/hip_bf16.h>

// ---------------- problem dims (match reference) ----------------
constexpr int M_DIM = 8192;    // tokens
constexpr int K_DIM = 4096;    // d_in
constexpr int N_DIM = 16384;   // d_out

// ---------------- tiling ----------------
constexpr int BM = 128;
constexpr int BN = 128;
constexpr int BK = 64;               // two WMMA K-steps of 32
constexpr int LDT = BK + 8;          // padded LDS row stride (bf16 elems), 16B-aligned
constexpr int TILE = BM * LDT;       // elems per LDS tile buffer
constexpr size_t SMEM_BYTES = 4ull * TILE * sizeof(unsigned short); // A x2 + B x2 = 72KB

typedef __bf16 v16bf __attribute__((ext_vector_type(16)));
typedef float  v8f   __attribute__((ext_vector_type(8)));
typedef int    v4i   __attribute__((ext_vector_type(4)));

#define GAS __attribute__((address_space(1)))
#define LAS __attribute__((address_space(3)))

#if __has_builtin(__builtin_amdgcn_global_load_async_to_lds_b128)
#define HAVE_ASYNC_LDS 1
#else
#define HAVE_ASYNC_LDS 0
#endif

#if __has_builtin(__builtin_amdgcn_cvt_pk_f32_fp8)
#define HAVE_HW_FP8CVT 1
#else
#define HAVE_HW_FP8CVT 0
#endif

union FragU { uint4 q[2]; v16bf v; };

// Pack two f32 to bf16x2 with round-to-nearest-even.
__device__ __forceinline__ unsigned pack_bf16_rne(float a, float b) {
#if __has_builtin(__builtin_amdgcn_cvt_pk_bf16_f32)
    typedef __bf16 v2bf_t __attribute__((ext_vector_type(2)));
    v2bf_t t = __builtin_amdgcn_cvt_pk_bf16_f32(a, b);
    return __builtin_bit_cast(unsigned, t);
#else
    unsigned u0 = __float_as_uint(a), u1 = __float_as_uint(b);
    u0 += 0x7FFFu + ((u0 >> 16) & 1u);
    u1 += 0x7FFFu + ((u1 >> 16) & 1u);
    return (u0 >> 16) | (u1 & 0xFFFF0000u);
#endif
}

// Exact manual fp8(e4m3)->f32 with scale folded in:
//   f32 = as_float(sign<<31 | (exp:mant)<<20) * (2^120 * scale)
// Handles e4m3 denormals/zero exactly (f32 denorms enabled on gfx1250 kernels).
__device__ __forceinline__ float fp8_manual(unsigned b, float s) {
    return __uint_as_float(((b & 0x80u) << 24) | ((b & 0x7Fu) << 20)) * s;
}

__device__ __forceinline__ void wait_async_lds() {
#if HAVE_ASYNC_LDS
#if __has_builtin(__builtin_amdgcn_s_wait_asynccnt)
    __builtin_amdgcn_s_wait_asynccnt(0);
#else
    asm volatile("s_wait_asynccnt 0x0" ::: "memory");
#endif
#endif
}

__global__ __launch_bounds__(256)
void fp8_linear_wmma_kernel(const __hip_bfloat16* __restrict__ x,
                            const unsigned char*  __restrict__ w,     // fp8 e4m3 [N,K]
                            const float*          __restrict__ scale, // per-tensor
                            const __hip_bfloat16* __restrict__ bias,
                            __hip_bfloat16*       __restrict__ out) {
    extern __shared__ __align__(16) unsigned short smem[];
    unsigned short* sA = smem;              // two BMxBK bf16 buffers
    unsigned short* sB = smem + 2 * TILE;   // two BNxBK bf16 buffers (dequantized W)

    const int tid  = threadIdx.x;
    const int lane = tid & 31;
    const int wid  = tid >> 5;
    const int wm   = wid & 3;     // wave M slot: rows wm*32 .. wm*32+31
    const int wn   = wid >> 2;    // wave N slot: cols wn*64 .. wn*64+63
    const int bm   = blockIdx.y * BM;
    const int bn   = blockIdx.x * BN;

    const float sc_raw = scale[0];
    const float sc_man = sc_raw * 0x1p+120f;   // for manual path

    const uint4* xg = reinterpret_cast<const uint4*>(x);   // row stride K/8 uint4 (8 bf16 each)
    const uint4* wg = reinterpret_cast<const uint4*>(w);   // row stride K/16 uint4 (16 fp8 each)

    // ---- stage helpers -----------------------------------------------------
    // A tile: 128x64 bf16 = 16KB -> 4 x 16B per thread. Via async LDS DMA when
    // available (no VGPR staging, no spills); else prefetch early + copy late.
    // W tile: 128x64 fp8 = 8KB -> 2 x 16B per thread; dequant in regs (8 VGPRs).

    auto early_stage = [&](int buf, int k0) {
#if HAVE_ASYNC_LDS
        unsigned short* a = sA + buf * TILE;
        #pragma unroll
        for (int i = 0; i < 4; ++i) {
            int c = tid + i * 256;
            int r = c >> 3, cc = c & 7;
            const uint4* gp = xg + (size_t)(bm + r) * (K_DIM / 8) + (k0 >> 3) + cc;
            __builtin_amdgcn_global_load_async_to_lds_b128(
                (GAS v4i*)gp, (LAS v4i*)(a + r * LDT + cc * 8), 0, 0);
        }
#else
        (void)buf;
        #pragma unroll
        for (int i = 0; i < 4; ++i) {
            int c = tid + i * 256;
            int r = c >> 3, cc = c & 7;
            __builtin_prefetch(xg + (size_t)(bm + r) * (K_DIM / 8) + (k0 >> 3) + cc, 0, 1);
        }
#endif
        #pragma unroll
        for (int i = 0; i < 2; ++i) {
            int c = tid + i * 256;
            int r = c >> 2, cc = c & 3;
            __builtin_prefetch(wg + (size_t)(bn + r) * (K_DIM / 16) + (k0 >> 4) + cc, 0, 1);
        }
    };

    auto late_stage = [&](int buf, int k0) {
#if !HAVE_ASYNC_LDS
        {   // sync A copy: load + immediately store (short live range, cache-hot)
            unsigned short* a = sA + buf * TILE;
            #pragma unroll
            for (int i = 0; i < 4; ++i) {
                int c = tid + i * 256;
                int r = c >> 3, cc = c & 7;
                uint4 v = xg[(size_t)(bm + r) * (K_DIM / 8) + (k0 >> 3) + cc];
                *reinterpret_cast<uint4*>(a + r * LDT + cc * 8) = v;
            }
        }
#endif
        unsigned short* b = sB + buf * TILE;
        #pragma unroll
        for (int i = 0; i < 2; ++i) {
            int c = tid + i * 256;
            int r = c >> 2, cc = c & 3;
            const uint4 raw = wg[(size_t)(bn + r) * (K_DIM / 16) + (k0 >> 4) + cc];
            const unsigned rr[4] = { raw.x, raw.y, raw.z, raw.w };
            unsigned o[8];
            #pragma unroll
            for (int j = 0; j < 4; ++j) {
#if HAVE_HW_FP8CVT
                auto lo = __builtin_amdgcn_cvt_pk_f32_fp8((int)rr[j], false); // bytes 0,1
                auto hi = __builtin_amdgcn_cvt_pk_f32_fp8((int)rr[j], true);  // bytes 2,3
                o[2 * j]     = pack_bf16_rne(lo[0] * sc_raw, lo[1] * sc_raw);
                o[2 * j + 1] = pack_bf16_rne(hi[0] * sc_raw, hi[1] * sc_raw);
#else
                float f0 = fp8_manual( rr[j]        & 0xFFu, sc_man);
                float f1 = fp8_manual((rr[j] >>  8) & 0xFFu, sc_man);
                float f2 = fp8_manual((rr[j] >> 16) & 0xFFu, sc_man);
                float f3 = fp8_manual( rr[j] >> 24        , sc_man);
                o[2 * j]     = pack_bf16_rne(f0, f1);
                o[2 * j + 1] = pack_bf16_rne(f2, f3);
#endif
            }
            *reinterpret_cast<uint4*>(b + r * LDT + cc * 16)     = make_uint4(o[0], o[1], o[2], o[3]);
            *reinterpret_cast<uint4*>(b + r * LDT + cc * 16 + 8) = make_uint4(o[4], o[5], o[6], o[7]);
        }
    };

    // ---- accumulators ------------------------------------------------------
    v8f acc[2][4];
    #pragma unroll
    for (int mi = 0; mi < 2; ++mi)
        #pragma unroll
        for (int ni = 0; ni < 4; ++ni)
            acc[mi][ni] = (v8f){0.f, 0.f, 0.f, 0.f, 0.f, 0.f, 0.f, 0.f};

    const int lo16 = lane & 15;
    const int hsel = lane >> 4;

    auto compute = [&](int buf) {
        const unsigned short* a = sA + buf * TILE;
        const unsigned short* b = sB + buf * TILE;
        #pragma unroll
        for (int ks = 0; ks < 2; ++ks) {
            const int ko = ks * 32;
            v16bf af[2], bfv[4];
            // A 16x32 bf16 frag: low lanes halves0-7=K0-7, halves8-15=K16-23;
            // high lanes +8 in K. -> two ds_load_b128 at +0 and +16 elems.
            #pragma unroll
            for (int mi = 0; mi < 2; ++mi) {
                const unsigned short* p = a + (wm * 32 + mi * 16 + lo16) * LDT + ko + hsel * 8;
                FragU u;
                u.q[0] = *reinterpret_cast<const uint4*>(p);
                u.q[1] = *reinterpret_cast<const uint4*>(p + 16);
                af[mi] = u.v;
            }
            // B 32x16 bf16 frag: lane = column N; same K order as A (K0-7,K16-23; high lanes +8).
            #pragma unroll
            for (int ni = 0; ni < 4; ++ni) {
                const unsigned short* p = b + (wn * 64 + ni * 16 + lo16) * LDT + ko + hsel * 8;
                FragU u;
                u.q[0] = *reinterpret_cast<const uint4*>(p);
                u.q[1] = *reinterpret_cast<const uint4*>(p + 16);
                bfv[ni] = u.v;
            }
            #pragma unroll
            for (int mi = 0; mi < 2; ++mi)
                #pragma unroll
                for (int ni = 0; ni < 4; ++ni)
                    acc[mi][ni] = __builtin_amdgcn_wmma_f32_16x16x32_bf16(
                        false, af[mi], false, bfv[ni], (short)0, acc[mi][ni], false, false);
        }
    };

    // ---- pipeline ----------------------------------------------------------
    early_stage(0, 0);
    late_stage(0, 0);
    wait_async_lds();
    __syncthreads();

    constexpr int KITERS = K_DIM / BK;  // 64
    for (int t = 0; t < KITERS; ++t) {
        const int buf = t & 1;
        if (t + 1 < KITERS) early_stage(buf ^ 1, (t + 1) * BK);  // async A / prefetch
        compute(buf);                                            // 16 WMMAs
        if (t + 1 < KITERS) late_stage(buf ^ 1, (t + 1) * BK);   // W dequant -> LDS
        wait_async_lds();
        __syncthreads();
    }

    // ---- epilogue: bias + bf16 store ---------------------------------------
    #pragma unroll
    for (int mi = 0; mi < 2; ++mi) {
        const int row0 = bm + wm * 32 + mi * 16 + hsel * 8;  // C layout: lanes16-31 -> M+8
        #pragma unroll
        for (int ni = 0; ni < 4; ++ni) {
            const int col = bn + wn * 64 + ni * 16 + lo16;   // C layout: N = lane&15
            const float bv = __bfloat162float(bias[col]);
            #pragma unroll
            for (int v = 0; v < 8; ++v) {
                float r = acc[mi][ni][v] + bv;
                out[(size_t)(row0 + v) * N_DIM + col] = __float2bfloat16(r);
            }
        }
    }
}

extern "C" void kernel_launch(void* const* d_in, const int* in_sizes, int n_in,
                              void* d_out, int out_size, void* d_ws, size_t ws_size,
                              hipStream_t stream) {
    (void)in_sizes; (void)n_in; (void)out_size; (void)d_ws; (void)ws_size;
    const __hip_bfloat16* x     = (const __hip_bfloat16*)d_in[0];
    const unsigned char*  w     = (const unsigned char*)d_in[1];
    const float*          scale = (const float*)d_in[2];
    const __hip_bfloat16* bias  = (const __hip_bfloat16*)d_in[3];
    __hip_bfloat16*       out   = (__hip_bfloat16*)d_out;

    (void)hipFuncSetAttribute((const void*)fp8_linear_wmma_kernel,
                              hipFuncAttributeMaxDynamicSharedMemorySize, (int)SMEM_BYTES);

    dim3 grid(N_DIM / BN, M_DIM / BM);   // 128 x 64 workgroups
    dim3 block(256, 1, 1);               // 8 wave32s
    fp8_linear_wmma_kernel<<<grid, block, SMEM_BYTES, stream>>>(x, w, scale, bias, out);
}
